// AdaptiveHyperModalityLayer_45311904973235
// MI455X (gfx1250) — matrix operand order repaired and checked by reading, not compiled, not run on hardware
//
#include <hip/hip_runtime.h>
#include <hip/hip_bf16.h>

typedef __bf16 bf16_t;
typedef __attribute__((ext_vector_type(16))) __bf16 v16bf;
typedef __attribute__((ext_vector_type(8)))  __bf16 v8bf;
typedef __attribute__((ext_vector_type(8)))  float  v8f;

// ---------------------------------------------------------------------------
// WMMA helper: D = A(16x32 bf16) * B(32x16 bf16) + C(16x16 f32)
// ---------------------------------------------------------------------------
__device__ inline v8f wmma_bf16(v16bf a, v16bf b, v8f c) {
  return __builtin_amdgcn_wmma_f32_16x16x32_bf16(
      /*neg_a=*/false, a, /*neg_b=*/false, b,
      /*c_mod=*/(short)0, c, /*reuse_a=*/false, /*reuse_b=*/false);
}

// ---------------------------------------------------------------------------
// CDNA5 async global->LDS copy (tracked by ASYNCcnt). No builtin on ROCm 7.2,
// so use inline asm per CDNA5 ISA / CDNA5 ISA (op 98).
// vdst VGPR carries the per-lane LDS byte offset (= low 32 bits of the generic
// address, per the flat-aperture rule), vaddr carries the 64-bit global addr.
// ---------------------------------------------------------------------------
__device__ inline void async_load_b128_to_lds(const void* gptr, void* lptr) {
  unsigned lds_off = (unsigned)(uintptr_t)lptr;
  unsigned long long gaddr = (unsigned long long)(uintptr_t)gptr;
  asm volatile("global_load_async_to_lds_b128 %0, %1, off"
               :
               : "v"(lds_off), "v"(gaddr)
               : "memory");
}
__device__ inline void wait_async0() {
  asm volatile("s_wait_asynccnt 0x0" ::: "memory");
}

// A-fragment (16x32, bf16) for row m of a row-major [16+][K] bf16 matrix.
// CDNA5 layout: lanes 0-15 -> M=lane, K = k0+0..7 (v0-3) and k0+16..23 (v4-7)
//               lanes 16-31 -> M=lane-16, K = k0+8..15 and k0+24..31
__device__ inline v16bf load_a16(const bf16_t* base, int K, int m, int k0, int lh) {
  const v8bf lo = *(const v8bf*)(base + (long long)m * K + k0 + lh * 8);
  const v8bf hi = *(const v8bf*)(base + (long long)m * K + k0 + 16 + lh * 8);
  v16bf a;
#pragma unroll
  for (int i = 0; i < 8; ++i) { a[i] = lo[i]; a[i + 8] = hi[i]; }
  return a;
}

// ---------------------------------------------------------------------------
// Elementwise f32 -> bf16 convert
// ---------------------------------------------------------------------------
__global__ __launch_bounds__(256) void cvt_bf16_kernel(const float* __restrict__ x,
                                                       bf16_t* __restrict__ y,
                                                       long long n) {
  long long i = (long long)blockIdx.x * blockDim.x + threadIdx.x;
  if (i < n) y[i] = (bf16_t)x[i];
}

// w[Kdim][Ndim] f32  ->  wt[Ndim][Kdim] bf16  (weights: small, naive is fine)
__global__ __launch_bounds__(256) void transpose_cvt_kernel(const float* __restrict__ w,
                                                            bf16_t* __restrict__ wt,
                                                            int Kdim, int Ndim) {
  int n = blockIdx.x * blockDim.x + threadIdx.x;
  int k = blockIdx.y;
  if (n < Ndim) wt[(long long)n * Kdim + k] = (bf16_t)w[(long long)k * Ndim + n];
}

// Batched LDS-tiled bf16 transpose: src[z][R][C] -> dst[z][C][R].
// 32x32 tiles through padded LDS so both sides are fully coalesced.
__global__ __launch_bounds__(256) void transpose_bf16_kernel(
    const bf16_t* __restrict__ src, bf16_t* __restrict__ dst, int R, int C) {
  __shared__ bf16_t tile[32][33];
  const long long base = (long long)blockIdx.z * R * C;
  const int c0 = blockIdx.x * 32, r0 = blockIdx.y * 32;
  const int tx = threadIdx.x & 31, ty = threadIdx.x >> 5;  // 32 x 8
#pragma unroll
  for (int i = 0; i < 32; i += 8)
    tile[ty + i][tx] = src[base + (long long)(r0 + ty + i) * C + c0 + tx];
  __syncthreads();
#pragma unroll
  for (int i = 0; i < 32; i += 8)
    dst[base + (long long)(c0 + ty + i) * R + r0 + tx] = tile[tx][ty + i];
}

// ---------------------------------------------------------------------------
// Fused GEMM + bias (+ residual) + LayerNorm.
//   C[M][1024] = A_bf16[M][K] @ Bt_bf16[1024][K]^T + bias
//   optional: rawOut = C pre-LN (bf16, linear layout; used for V)
//   out = LN(C (+ resid)) * gamma + beta   (bf16 or f32)
// Workgroup: 256 threads = 8 waves; 16 rows x 1024 cols per block.
// Wave w owns cols [w*128, w*128+128) -> 8 C fragments.
// ---------------------------------------------------------------------------
template <bool RESIDUAL, bool RAW_OUT, bool F32OUT>
__global__ __launch_bounds__(256) void gemm_ln_kernel(
    const bf16_t* __restrict__ A, const bf16_t* __restrict__ Bt,
    const float* __restrict__ bias, const float* __restrict__ gamma,
    const float* __restrict__ beta, const float* __restrict__ resid,
    bf16_t* __restrict__ outB, float* __restrict__ outF,
    bf16_t* __restrict__ rawOut, int M, int K, int N) {
  const int tid = threadIdx.x;
  const int wave = tid >> 5, lane = tid & 31;
  const int lh = lane >> 4, ln = lane & 15;
  const long long m0 = (long long)blockIdx.x * 16;

  __shared__ __align__(32) bf16_t sA[16 * 1024];
  __shared__ float rsum[16 * 8];
  __shared__ float rsq[16 * 8];
  __shared__ float rmean[16];
  __shared__ float rrstd[16];

  // Async-stage the 16 x K A-tile straight into LDS (ASYNCcnt path).
  const int elems = 16 * K;
  for (int idx = tid * 8; idx < elems; idx += 256 * 8)
    async_load_b128_to_lds(A + m0 * K + idx, sA + idx);
  wait_async0();
  __syncthreads();

  v8f zero = {};
  v8f acc[8];
#pragma unroll
  for (int n = 0; n < 8; ++n) acc[n] = zero;

  const int colBase = wave * 128;
  for (int k0 = 0; k0 < K; k0 += 32) {
    v16bf a = load_a16(sA, K, ln, k0, lh);
#pragma unroll
    for (int n = 0; n < 8; ++n) {
      // B fragment: lane holds col (n-dim) = colBase+n*16+ln, k = k0+8h..+7 and k0+16+8h..+7.
      const bf16_t* bp = Bt + (long long)(colBase + n * 16 + ln) * K + k0 + lh * 8;
      const v8bf blo = *(const v8bf*)bp;
      const v8bf bhi = *(const v8bf*)(bp + 16);
      v16bf bfrag;
#pragma unroll
      for (int i = 0; i < 8; ++i) { bfrag[i] = blo[i]; bfrag[i + 8] = bhi[i]; }
      acc[n] = wmma_bf16(a, bfrag, acc[n]);
    }
  }

  // Epilogue: bias, optional residual / raw store, LN partials.
  float s1[8], s2[8];
#pragma unroll
  for (int i = 0; i < 8; ++i) { s1[i] = 0.f; s2[i] = 0.f; }

#pragma unroll
  for (int n = 0; n < 8; ++n) {
    const int col = colBase + n * 16 + ln;
    const float bcol = bias[col];
#pragma unroll
    for (int i = 0; i < 8; ++i) {
      const int m = i + lh * 8;
      float v = acc[n][i] + bcol;
      if constexpr (RESIDUAL) v += resid[(m0 + m) * N + col];
      if constexpr (RAW_OUT) rawOut[(m0 + m) * N + col] = (bf16_t)v;
      acc[n][i] = v;
      s1[i] += v;
      s2[i] += v * v;
    }
  }
  // Reduce across the 16 lanes of each half (xor masks < 16 stay in half).
#pragma unroll
  for (int off = 1; off < 16; off <<= 1) {
#pragma unroll
    for (int i = 0; i < 8; ++i) {
      s1[i] += __shfl_xor(s1[i], off, 32);
      s2[i] += __shfl_xor(s2[i], off, 32);
    }
  }
  if (ln == 0) {
#pragma unroll
    for (int i = 0; i < 8; ++i) {
      const int m = i + lh * 8;
      rsum[m * 8 + wave] = s1[i];
      rsq[m * 8 + wave] = s2[i];
    }
  }
  __syncthreads();
  if (tid < 16) {
    float a1 = 0.f, a2 = 0.f;
    for (int w2 = 0; w2 < 8; ++w2) { a1 += rsum[tid * 8 + w2]; a2 += rsq[tid * 8 + w2]; }
    const float mean = a1 / (float)N;
    const float var = a2 / (float)N - mean * mean;
    rmean[tid] = mean;
    rrstd[tid] = rsqrtf(var + 1e-5f);
  }
  __syncthreads();

#pragma unroll
  for (int n = 0; n < 8; ++n) {
    const int col = colBase + n * 16 + ln;
    const float g = gamma[col], bt = beta[col];
#pragma unroll
    for (int i = 0; i < 8; ++i) {
      const int m = i + lh * 8;
      const float v = (acc[n][i] - rmean[m]) * rrstd[m] * g + bt;
      const long long o = (m0 + m) * N + col;
      if constexpr (F32OUT) outF[o] = v; else outB[o] = (bf16_t)v;
    }
  }
}

// ---------------------------------------------------------------------------
// Attention: one workgroup = 16 query rows (one batch). Whole 16x2048 f32
// score strip + 16x1024 bf16 Q tile live in LDS (<= 320 KB/WGP on CDNA5),
// enabling exact single-pass softmax. Then H_hyper = alpha @ V via WMMA with
// V pre-transposed so B fragments are contiguous.
// ---------------------------------------------------------------------------
__global__ __launch_bounds__(256) void attn_kernel(
    const bf16_t* __restrict__ Qb, const bf16_t* __restrict__ Kb,
    const bf16_t* __restrict__ Vt, bf16_t* __restrict__ Hh,
    int L, int S, int H) {
  extern __shared__ char smem[];
  bf16_t* sQ = (bf16_t*)smem;                              // 16*H bf16   (32 KB)
  float* sS = (float*)(smem + 16 * 1024 * 2);              // 16*S f32   (128 KB)
  float* red = (float*)(smem + 16 * 1024 * 2 + 16 * 2048 * 4);
  float* red2 = red + 256;
  float* rowmax = red2 + 256;
  float* rowsum = rowmax + 16;

  const int tid = threadIdx.x;
  const int wave = tid >> 5, lane = tid & 31;
  const int lh = lane >> 4, ln = lane & 15;
  const long long row0 = (long long)blockIdx.x * 16;
  const int b = (int)(row0 / L);
  const bf16_t* Kbb = Kb + (long long)b * S * H;
  const bf16_t* Vtb = Vt + (long long)b * H * S;

  // Phase 1: Q tile -> LDS via async DMA path.
  for (int idx = tid * 8; idx < 16 * H; idx += 256 * 8)
    async_load_b128_to_lds(Qb + row0 * H + idx, sQ + idx);
  wait_async0();
  __syncthreads();

  // Phase 2: scores = scale * Q @ K^T. Wave w owns score cols [w*256, w*256+256).
  const float scale = 0.03125f;  // 1/sqrt(1024)
  const int sBase = wave * 256;
  for (int c = 0; c < 16; ++c) {
    v8f facc = {};
    const int scol = sBase + c * 16 + ln;
    const bf16_t* kp = Kbb + (long long)scol * H + lh * 8;
    for (int k0 = 0; k0 < H; k0 += 32) {
      v16bf a = load_a16(sQ, H, ln, k0, lh);
      const v8bf blo = *(const v8bf*)(kp + k0);
      const v8bf bhi = *(const v8bf*)(kp + k0 + 16);
      v16bf bfrag;
#pragma unroll
      for (int i = 0; i < 8; ++i) { bfrag[i] = blo[i]; bfrag[i + 8] = bhi[i]; }
      facc = wmma_bf16(a, bfrag, facc);
    }
#pragma unroll
    for (int i = 0; i < 8; ++i) {
      const int m = i + lh * 8;
      sS[m * S + scol] = facc[i] * scale;
    }
  }
  __syncthreads();

  // Phase 3: exact softmax on the full 16 x S strip. 16 threads per row.
  const int r = tid & 15, g = tid >> 4;
  {
    const float* rp = sS + r * S + g * 128;
    float mx = rp[0];
    for (int j = 1; j < 128; ++j) mx = fmaxf(mx, rp[j]);
    red[g * 16 + r] = mx;
  }
  __syncthreads();
  if (tid < 16) {
    float mx = red[tid];
    for (int gg = 1; gg < 16; ++gg) mx = fmaxf(mx, red[gg * 16 + tid]);
    rowmax[tid] = mx;
  }
  __syncthreads();
  {
    const float mx = rowmax[r];
    float* rp = sS + r * S + g * 128;
    float sum = 0.f;
    for (int j = 0; j < 128; ++j) {
      const float e = __expf(rp[j] - mx);
      rp[j] = e;
      sum += e;
    }
    red2[g * 16 + r] = sum;
  }
  __syncthreads();
  if (tid < 16) {
    float s = 0.f;
    for (int gg = 0; gg < 16; ++gg) s += red2[gg * 16 + tid];
    rowsum[tid] = 1.0f / s;
  }
  __syncthreads();

  // Phase 4: H_hyper = p @ V. Wave w owns output cols [w*128, w*128+128).
  v8f zero = {};
  v8f acc[8];
#pragma unroll
  for (int n = 0; n < 8; ++n) acc[n] = zero;
  const int h0 = wave * 128;
  for (int ks = 0; ks < S; ks += 32) {
    // A fragment from LDS scores (f32 -> bf16 at fragment build).
    const float* p0 = sS + ln * S + ks + lh * 8;
    const float* p1 = p0 + 16;
    v16bf a;
#pragma unroll
    for (int i = 0; i < 8; ++i) { a[i] = (bf16_t)p0[i]; a[i + 8] = (bf16_t)p1[i]; }
#pragma unroll
    for (int n = 0; n < 8; ++n) {
      const bf16_t* vp = Vtb + (long long)(h0 + n * 16 + ln) * S + ks + lh * 8;
      const v8bf blo = *(const v8bf*)vp;
      const v8bf bhi = *(const v8bf*)(vp + 16);
      v16bf bfrag;
#pragma unroll
      for (int i = 0; i < 8; ++i) { bfrag[i] = blo[i]; bfrag[i + 8] = bhi[i]; }
      acc[n] = wmma_bf16(a, bfrag, acc[n]);
    }
  }

  // Phase 5: normalize rows by 1/sum and store bf16 H_hyper.
#pragma unroll
  for (int n = 0; n < 8; ++n) {
    const int col = h0 + n * 16 + ln;
#pragma unroll
    for (int i = 0; i < 8; ++i) {
      const int m = i + lh * 8;
      Hh[(row0 + m) * H + col] = (bf16_t)(acc[n][i] * rowsum[m]);
    }
  }
}

// ---------------------------------------------------------------------------
// Host launch
// ---------------------------------------------------------------------------
extern "C" void kernel_launch(void* const* d_in, const int* in_sizes, int n_in,
                              void* d_out, int out_size, void* d_ws, size_t ws_size,
                              hipStream_t stream) {
  const float* H_l = (const float*)d_in[0];
  const float* H_a = (const float*)d_in[1];
  const float* W_text = (const float*)d_in[2];
  const float* b_text = (const float*)d_in[3];
  const float* W_audio = (const float*)d_in[4];
  const float* b_audio = (const float*)d_in[5];
  const float* W_out = (const float*)d_in[6];
  const float* b_out = (const float*)d_in[7];
  const float* g1 = (const float*)d_in[8];
  const float* beta1 = (const float*)d_in[9];
  const float* g2 = (const float*)d_in[10];
  const float* beta2 = (const float*)d_in[11];
  const float* g_out = (const float*)d_in[12];
  const float* beta_out = (const float*)d_in[13];

  const int B = 16, L = 1024, S = 2048, DT = 1024, DA = 768, H = 1024;
  const long long ML = (long long)B * L;   // 16384 text rows
  const long long MS = (long long)B * S;   // 32768 audio rows

  char* ws = (char*)d_ws;
  size_t off = 0;
  auto wsalloc = [&](size_t bytes) -> char* {
    char* p = ws + off;
    off += (bytes + 255) & ~(size_t)255;
    return p;
  };
  bf16_t* Hlb = (bf16_t*)wsalloc((size_t)ML * DT * 2);       // H_l bf16
  bf16_t* Hab = (bf16_t*)wsalloc((size_t)MS * DA * 2);       // H_a bf16
  bf16_t* WtT = (bf16_t*)wsalloc((size_t)H * DT * 2);        // W_text^T
  bf16_t* WaT = (bf16_t*)wsalloc((size_t)H * DA * 2);        // W_audio^T
  bf16_t* WoT = (bf16_t*)wsalloc((size_t)H * H * 2);         // W_out^T
  bf16_t* Qb = (bf16_t*)wsalloc((size_t)ML * H * 2);         // LN(text proj)
  bf16_t* Kb = (bf16_t*)wsalloc((size_t)MS * H * 2);         // LN(audio proj)
  bf16_t* Vb = (bf16_t*)wsalloc((size_t)MS * H * 2);         // raw audio proj (V)
  bf16_t* Vt = (bf16_t*)wsalloc((size_t)B * H * S * 2);      // V transposed [b][h][s]
  bf16_t* Hh = (bf16_t*)wsalloc((size_t)ML * H * 2);         // H_hyper

  // Precision staging: f32 -> bf16, weights transposed for contiguous B-frags.
  {
    long long n = ML * DT;
    cvt_bf16_kernel<<<dim3((unsigned)((n + 255) / 256)), 256, 0, stream>>>(H_l, Hlb, n);
  }
  {
    long long n = MS * DA;
    cvt_bf16_kernel<<<dim3((unsigned)((n + 255) / 256)), 256, 0, stream>>>(H_a, Hab, n);
  }
  transpose_cvt_kernel<<<dim3(4, DT), 256, 0, stream>>>(W_text, WtT, DT, H);
  transpose_cvt_kernel<<<dim3(4, DA), 256, 0, stream>>>(W_audio, WaT, DA, H);
  transpose_cvt_kernel<<<dim3(4, H), 256, 0, stream>>>(W_out, WoT, H, H);

  // Q = LN(H_l @ W_text + b_text)
  gemm_ln_kernel<false, false, false><<<dim3((unsigned)(ML / 16)), 256, 0, stream>>>(
      Hlb, WtT, b_text, g1, beta1, nullptr, Qb, nullptr, nullptr, (int)ML, DT, H);

  // K = LN(H_a @ W_audio + b_audio); V = raw projection (linear, coalesced).
  gemm_ln_kernel<false, true, false><<<dim3((unsigned)(MS / 16)), 256, 0, stream>>>(
      Hab, WaT, b_audio, g2, beta2, nullptr, Kb, nullptr, Vb, (int)MS, DA, H);

  // Vb[b][s][h] -> Vt[b][h][s] via coalesced LDS-tiled transpose.
  transpose_bf16_kernel<<<dim3(H / 32, S / 32, B), 256, 0, stream>>>(Vb, Vt, S, H);

  // Attention: softmax(QK^T/sqrt(H)) @ V
  const size_t smemAttn =
      (size_t)16 * 1024 * 2 + (size_t)16 * 2048 * 4 + (256 + 256 + 16 + 16) * 4;
  attn_kernel<<<dim3((unsigned)(ML / 16)), 256, smemAttn, stream>>>(
      Qb, Kb, Vt, Hh, L, S, H);

  // out = LN(H_hyper @ W_out + b_out + H_l)
  gemm_ln_kernel<true, false, true><<<dim3((unsigned)(ML / 16)), 256, 0, stream>>>(
      Hh, WoT, b_out, g_out, beta_out, H_l, nullptr, (float*)d_out, nullptr,
      (int)ML, H, H);
}
